// GraphConvBlock_15040975470647
// MI455X (gfx1250) — matrix operand order repaired and checked by reading, not compiled, not run on hardware
//
#include <hip/hip_runtime.h>
#include <hip/hip_bf16.h>

typedef __attribute__((ext_vector_type(16))) __bf16 v16bf;
typedef __attribute__((ext_vector_type(8)))  float  v8f;

#define ND 256
#define ED 128
#define EH 256
#define K1DIM 640  // 2*ND + ED

#define CSTR 656   // comb LDS stride (bf16 elems), mult of 8
#define HSTR 264   // hidden LDS stride
#define USTR 132   // upd f32 LDS stride (edge)
#define ESTR 136   // e-tile LDS stride
#define MSTR 264   // messages LDS stride
#define NUSTR 260  // node upd f32 stride

// ---------- WMMA fragment helpers (layouts per CDNA5 ISA 7.12.2) ----------

__device__ __forceinline__ void st4bf(__bf16* dst, float4 v) {
  union { __bf16 h[4]; unsigned long long u; } t;
  t.h[0] = (__bf16)v.x; t.h[1] = (__bf16)v.y;
  t.h[2] = (__bf16)v.z; t.h[3] = (__bf16)v.w;
  *(unsigned long long*)dst = t.u;
}

// A fragment 16x32 bf16 from row-major LDS tile. lane<16: K=kb+[0..7], kb+[16..23];
// lane>=16: K=kb+[8..15], kb+[24..31]; row = lane%16.
__device__ __forceinline__ v16bf load_a(const __bf16* tile, int stride, int row, int half, int kb) {
  const __bf16* p0 = tile + row * stride + kb + half * 8;
  const __bf16* p1 = p0 + 16;
  v16bf a;
#pragma unroll
  for (int i = 0; i < 8; ++i) { a[i] = p0[i]; a[i + 8] = p1[i]; }
  return a;
}

// B fragment 32x16 bf16 from transposed weights Wt (N rows x K cols, row-major).
// lane<16: col=nb+lane, K=kb+[0..7], kb+[16..23] ; lane>=16: col=nb+lane-16, K=kb+[8..15], kb+[24..31].
__device__ __forceinline__ v16bf load_b(const __bf16* __restrict__ wt, int kdim, int nb, int kb, int lane) {
  const __bf16* p = wt + (size_t)(nb + (lane & 15)) * kdim + kb + ((lane >> 4) << 3);
  v16bf b;
#pragma unroll
  for (int i = 0; i < 8; ++i) { b[i] = p[i]; b[i + 8] = p[16 + i]; }
  return b;
}

__device__ __forceinline__ v8f wmma_bf16(v16bf a, v16bf b, v8f c) {
  return __builtin_amdgcn_wmma_f32_16x16x32_bf16(false, a, false, b, (short)0, c, false, false);
}

__device__ __forceinline__ void atomicMaxF32(float* addr, float val) {
  if (val >= 0.0f) atomicMax((int*)addr, __float_as_int(val));
  else             atomicMin((unsigned int*)addr, __float_as_uint(val));
}

// ---------- preprocessing kernels ----------

__global__ void k_transpose_bf16(const float* __restrict__ in, __bf16* __restrict__ out,
                                 int R, int C) {
  int idx = blockIdx.x * blockDim.x + threadIdx.x;
  if (idx < R * C) {
    int r = idx / C, c = idx - r * C;
    out[(size_t)c * R + r] = (__bf16)in[(size_t)r * C + c];
  }
}

__global__ void k_awm(const float* __restrict__ aw, const float* __restrict__ ab,
                      float* __restrict__ awm) {
  int i = blockIdx.x * blockDim.x + threadIdx.x;
  if (i < ND + ED)
    awm[i] = 0.25f * (aw[i * 4] + aw[i * 4 + 1] + aw[i * 4 + 2] + aw[i * 4 + 3]);
  else if (i == ND + ED)
    awm[ND + ED] = 0.25f * (ab[0] + ab[1] + ab[2] + ab[3]);
}

__global__ void k_init(float* __restrict__ mx, float* __restrict__ se,
                       float* __restrict__ msgs, int N) {
  int i = blockIdx.x * blockDim.x + threadIdx.x;
  if (i < N * ND) msgs[i] = 0.0f;
  if (i < N) { mx[i] = -__builtin_inff(); se[i] = 0.0f; }
}

// ---------- edge update: fused gather + MLP + LN + score + segment-max ----------

__global__ __launch_bounds__(32) void k_edge(
    const float* __restrict__ nf, const float* __restrict__ ef,
    const long long* __restrict__ ei,
    const __bf16* __restrict__ ew1t, const float* __restrict__ eb1,
    const __bf16* __restrict__ ew2t, const float* __restrict__ eb2,
    const float* __restrict__ eng, const float* __restrict__ enb,
    const float* __restrict__ awm,
    float* __restrict__ e_out, float* __restrict__ scores,
    float* __restrict__ mx, int E)
{
  __shared__ __attribute__((aligned(16))) __bf16 comb[16 * CSTR];
  __shared__ __attribute__((aligned(16))) __bf16 hid [16 * HSTR];
  __shared__ __attribute__((aligned(16))) float  upd [16 * USTR];

  int lane = threadIdx.x;
  int row  = lane & 15, half = lane >> 4, col = row;
  long long edge = (long long)blockIdx.x * 16 + row;
  long long sidx = ei[edge];
  long long didx = ei[(long long)E + edge];

  // gather [src | dst | e] as bf16 into LDS (2 lanes per row, 128/128/64 floats each)
  {
    const float4* s4 = (const float4*)(nf + (size_t)sidx * ND);
    const float4* d4 = (const float4*)(nf + (size_t)didx * ND);
    const float4* f4 = (const float4*)(ef + (size_t)edge * ED);
    __bf16* crow = comb + row * CSTR;
#pragma unroll
    for (int i = 0; i < 32; ++i) st4bf(crow + half * 128 + i * 4, s4[half * 32 + i]);
#pragma unroll
    for (int i = 0; i < 32; ++i) st4bf(crow + ND + half * 128 + i * 4, d4[half * 32 + i]);
#pragma unroll
    for (int i = 0; i < 16; ++i) st4bf(crow + 2 * ND + half * 64 + i * 4, f4[half * 16 + i]);
  }
  __syncthreads();

  // GEMM1: (16x640) @ (640x256), ReLU -> hidden bf16
  for (int nt = 0; nt < EH / 16; ++nt) {
    v8f acc = {0.f,0.f,0.f,0.f,0.f,0.f,0.f,0.f};
    for (int kt = 0; kt < K1DIM / 32; ++kt)
      acc = wmma_bf16(load_a(comb, CSTR, row, half, kt * 32),
                      load_b(ew1t, K1DIM, nt * 16, kt * 32, lane), acc);
    float bias = eb1[nt * 16 + col];
#pragma unroll
    for (int r = 0; r < 8; ++r) {
      float v = fmaxf(acc[r] + bias, 0.0f);
      hid[(r + half * 8) * HSTR + nt * 16 + col] = (__bf16)v;
    }
  }
  __syncthreads();

  // GEMM2: (16x256) @ (256x128) -> upd f32
  for (int nt = 0; nt < ED / 16; ++nt) {
    v8f acc = {0.f,0.f,0.f,0.f,0.f,0.f,0.f,0.f};
    for (int kt = 0; kt < EH / 32; ++kt)
      acc = wmma_bf16(load_a(hid, HSTR, row, half, kt * 32),
                      load_b(ew2t, EH, nt * 16, kt * 32, lane), acc);
    float bias = eb2[nt * 16 + col];
#pragma unroll
    for (int r = 0; r < 8; ++r)
      upd[(r + half * 8) * USTR + nt * 16 + col] = acc[r] + bias;
  }
  __syncthreads();

  // residual + LayerNorm + score + segment-max (one lane per edge row)
  if (lane < 16) {
    const float* efr = ef + (size_t)edge * ED;
    float* ur = upd + row * USTR;
    float s = 0.0f;
    for (int c = 0; c < ED; ++c) { float x = efr[c] + ur[c]; ur[c] = x; s += x; }
    float mean = s * (1.0f / ED);
    float var = 0.0f;
    for (int c = 0; c < ED; ++c) { float d = ur[c] - mean; var += d * d; }
    float inv = rsqrtf(var * (1.0f / ED) + 1e-5f);
    float* eo = e_out + (size_t)edge * ED;
    float sc = awm[ND + ED];
    for (int c = 0; c < ED; ++c) {
      float y = (ur[c] - mean) * inv * eng[c] + enb[c];
      eo[c] = y;
      sc += y * awm[ND + c];
    }
    const __bf16* drow = comb + row * CSTR + ND;  // gathered dst features
    for (int c = 0; c < ND; ++c) sc += (float)drow[c] * awm[c];
    scores[edge] = sc;
    atomicMaxF32(mx + didx, sc);
  }
}

// ---------- exp + segment-sum ----------

__global__ void k_exp(const float* __restrict__ scores, const long long* __restrict__ ei,
                      const float* __restrict__ mx, float* __restrict__ exw,
                      float* __restrict__ se, int E)
{
  long long i = (long long)blockIdx.x * blockDim.x + threadIdx.x;
  if (i < E) {
    long long d = ei[(long long)E + i];
    float ex = __expf(scores[i] - mx[d]);
    exw[i] = ex;
    atomicAdd(se + d, ex);
  }
}

// ---------- values GEMM + weighted scatter into messages ----------

__global__ __launch_bounds__(32) void k_scatter(
    const float* __restrict__ e_in, const long long* __restrict__ ei,
    const __bf16* __restrict__ evwt, const float* __restrict__ evb,
    const float* __restrict__ exw, const float* __restrict__ se,
    float* __restrict__ msgs, int E)
{
  __shared__ __attribute__((aligned(16))) __bf16 et[16 * ESTR];
  __shared__ float wrow[16];
  __shared__ int   drow[16];

  int lane = threadIdx.x;
  int row = lane & 15, half = lane >> 4, col = row;
  long long edge = (long long)blockIdx.x * 16 + row;

  const float4* e4 = (const float4*)(e_in + (size_t)edge * ED);
  __bf16* er = et + row * ESTR;
#pragma unroll
  for (int i = 0; i < 16; ++i) st4bf(er + half * 64 + i * 4, e4[half * 16 + i]);
  if (lane < 16) {
    long long d = ei[(long long)E + edge];
    drow[row] = (int)d;
    wrow[row] = exw[edge] / (se[d] + 1e-6f);
  }
  __syncthreads();

  // values = e @ evw (128->256), scaled scatter-add
  for (int nt = 0; nt < ND / 16; ++nt) {
    v8f acc = {0.f,0.f,0.f,0.f,0.f,0.f,0.f,0.f};
    for (int kt = 0; kt < ED / 32; ++kt)
      acc = wmma_bf16(load_a(et, ESTR, row, half, kt * 32),
                      load_b(evwt, ED, nt * 16, kt * 32, lane), acc);
    float bias = evb[nt * 16 + col];
#pragma unroll
    for (int r = 0; r < 8; ++r) {
      int rr = r + half * 8;
      atomicAdd(msgs + (size_t)drow[rr] * ND + nt * 16 + col, (acc[r] + bias) * wrow[rr]);
    }
  }
}

// ---------- node MLP + LN ----------

__global__ __launch_bounds__(32) void k_node(
    const float* __restrict__ nf, const float* __restrict__ msgs,
    const __bf16* __restrict__ ow1t, const float* __restrict__ ob1,
    const __bf16* __restrict__ ow2t, const float* __restrict__ ob2,
    const float* __restrict__ nng, const float* __restrict__ nnb,
    float* __restrict__ n_out, int N)
{
  __shared__ __attribute__((aligned(16))) __bf16 mt [16 * MSTR];
  __shared__ __attribute__((aligned(16))) __bf16 hid[16 * MSTR];
  __shared__ __attribute__((aligned(16))) float  upd[16 * NUSTR];

  int lane = threadIdx.x;
  int row = lane & 15, half = lane >> 4, col = row;
  int node = blockIdx.x * 16 + row;
  int nodec = node < N ? node : N - 1;

  const float4* m4 = (const float4*)(msgs + (size_t)nodec * ND);
  __bf16* mr = mt + row * MSTR;
#pragma unroll
  for (int i = 0; i < 32; ++i) st4bf(mr + half * 128 + i * 4, m4[half * 32 + i]);
  __syncthreads();

  for (int nt = 0; nt < ND / 16; ++nt) {
    v8f acc = {0.f,0.f,0.f,0.f,0.f,0.f,0.f,0.f};
    for (int kt = 0; kt < ND / 32; ++kt)
      acc = wmma_bf16(load_a(mt, MSTR, row, half, kt * 32),
                      load_b(ow1t, ND, nt * 16, kt * 32, lane), acc);
    float bias = ob1[nt * 16 + col];
#pragma unroll
    for (int r = 0; r < 8; ++r) {
      float v = fmaxf(acc[r] + bias, 0.0f);
      hid[(r + half * 8) * MSTR + nt * 16 + col] = (__bf16)v;
    }
  }
  __syncthreads();

  for (int nt = 0; nt < ND / 16; ++nt) {
    v8f acc = {0.f,0.f,0.f,0.f,0.f,0.f,0.f,0.f};
    for (int kt = 0; kt < ND / 32; ++kt)
      acc = wmma_bf16(load_a(hid, MSTR, row, half, kt * 32),
                      load_b(ow2t, ND, nt * 16, kt * 32, lane), acc);
    float bias = ob2[nt * 16 + col];
#pragma unroll
    for (int r = 0; r < 8; ++r)
      upd[(r + half * 8) * NUSTR + nt * 16 + col] = acc[r] + bias;
  }
  __syncthreads();

  if (lane < 16 && node < N) {
    const float* nr = nf + (size_t)node * ND;
    float* ur = upd + row * NUSTR;
    float s = 0.0f;
    for (int c = 0; c < ND; ++c) { float x = nr[c] + ur[c]; ur[c] = x; s += x; }
    float mean = s * (1.0f / ND);
    float var = 0.0f;
    for (int c = 0; c < ND; ++c) { float d = ur[c] - mean; var += d * d; }
    float inv = rsqrtf(var * (1.0f / ND) + 1e-5f);
    float* no = n_out + (size_t)node * ND;
    for (int c = 0; c < ND; ++c)
      no[c] = (ur[c] - mean) * inv * nng[c] + nnb[c];
  }
}

// ---------- launcher ----------

extern "C" void kernel_launch(void* const* d_in, const int* in_sizes, int n_in,
                              void* d_out, int out_size, void* d_ws, size_t ws_size,
                              hipStream_t stream) {
  const float*     nf  = (const float*)d_in[0];
  const float*     ef  = (const float*)d_in[1];
  const long long* ei  = (const long long*)d_in[2];
  const float* ew1 = (const float*)d_in[3];
  const float* eb1 = (const float*)d_in[4];
  const float* ew2 = (const float*)d_in[5];
  const float* eb2 = (const float*)d_in[6];
  const float* eng = (const float*)d_in[7];
  const float* enb = (const float*)d_in[8];
  const float* aw  = (const float*)d_in[9];
  const float* ab  = (const float*)d_in[10];
  const float* evw = (const float*)d_in[11];
  const float* evb = (const float*)d_in[12];
  const float* ow1 = (const float*)d_in[13];
  const float* ob1 = (const float*)d_in[14];
  const float* ow2 = (const float*)d_in[15];
  const float* ob2 = (const float*)d_in[16];
  const float* nng = (const float*)d_in[17];
  const float* nnb = (const float*)d_in[18];

  int N = in_sizes[0] / ND;   // 25000
  int E = in_sizes[1] / ED;   // 400000

  float* out_n = (float*)d_out;
  float* out_e = out_n + (size_t)N * ND;

  // workspace carve-up
  char* ws = (char*)d_ws;
  auto take = [&](size_t bytes) { char* p = ws; ws += (bytes + 255) & ~(size_t)255; return p; };
  __bf16* ew1t = (__bf16*)take((size_t)K1DIM * EH * 2);
  __bf16* ew2t = (__bf16*)take((size_t)EH * ED * 2);
  __bf16* evwt = (__bf16*)take((size_t)ED * ND * 2);
  __bf16* ow1t = (__bf16*)take((size_t)ND * ND * 2);
  __bf16* ow2t = (__bf16*)take((size_t)ND * ND * 2);
  float*  awm  = (float*)take((ND + ED + 1) * 4);
  float*  scor = (float*)take((size_t)E * 4);
  float*  exw  = (float*)take((size_t)E * 4);
  float*  mx   = (float*)take((size_t)N * 4);
  float*  se   = (float*)take((size_t)N * 4);
  float*  msgs = (float*)take((size_t)N * ND * 4);

  const int T = 256;
  k_transpose_bf16<<<(K1DIM * EH + T - 1) / T, T, 0, stream>>>(ew1, ew1t, K1DIM, EH);
  k_transpose_bf16<<<(EH * ED + T - 1) / T, T, 0, stream>>>(ew2, ew2t, EH, ED);
  k_transpose_bf16<<<(ED * ND + T - 1) / T, T, 0, stream>>>(evw, evwt, ED, ND);
  k_transpose_bf16<<<(ND * ND + T - 1) / T, T, 0, stream>>>(ow1, ow1t, ND, ND);
  k_transpose_bf16<<<(ND * ND + T - 1) / T, T, 0, stream>>>(ow2, ow2t, ND, ND);
  k_awm<<<2, T, 0, stream>>>(aw, ab, awm);
  k_init<<<(N * ND + T - 1) / T, T, 0, stream>>>(mx, se, msgs, N);

  k_edge<<<E / 16, 32, 0, stream>>>(nf, ef, ei, ew1t, eb1, ew2t, eb2,
                                    eng, enb, awm, out_e, scor, mx, E);
  k_exp<<<(E + T - 1) / T, T, 0, stream>>>(scor, ei, mx, exw, se, E);
  k_scatter<<<E / 16, 32, 0, stream>>>(out_e, ei, evwt, evb, exw, se, msgs, E);
  k_node<<<(N + 15) / 16, 32, 0, stream>>>(nf, msgs, ow1t, ob1, ow2t, ob2,
                                           nng, nnb, out_n, N);
}
